// GroupEncoder_9560597201434
// MI455X (gfx1250) — matrix operand order repaired and checked by reading, not compiled, not run on hardware
//
#include <hip/hip_runtime.h>
#include <hip/hip_bf16.h>

typedef __attribute__((ext_vector_type(16))) _Float16 v16h;
typedef __attribute__((ext_vector_type(8)))  _Float16 v8h;
typedef __attribute__((ext_vector_type(8)))  float    v8f;

#define NB 1000000   // reflections
#define NK 16384     // groups

__device__ __forceinline__ float silu_f(float v) {
    return v / (1.0f + __expf(-v));
}

// ---------------- threefry2x32 (key = jax.random.key(42) -> (0,42)) -------------
__device__ __forceinline__ unsigned rotl32(unsigned x, int r) {
    return (x << r) | (x >> (32 - r));
}

__device__ __forceinline__ void threefry2x32(unsigned c0, unsigned c1,
                                             unsigned& o0, unsigned& o1) {
    const unsigned k0 = 0u, k1 = 42u;
    const unsigned ks2 = k0 ^ k1 ^ 0x1BD11BDAu;
    unsigned x0 = c0 + k0, x1 = c1 + k1;
    const int RA[4] = {13, 15, 26, 6};
    const int RB[4] = {17, 29, 16, 24};
#pragma unroll
    for (int i = 0; i < 4; ++i) { x0 += x1; x1 = rotl32(x1, RA[i]); x1 ^= x0; }
    x0 += k1;  x1 += ks2 + 1u;
#pragma unroll
    for (int i = 0; i < 4; ++i) { x0 += x1; x1 = rotl32(x1, RB[i]); x1 ^= x0; }
    x0 += ks2; x1 += k0 + 2u;
#pragma unroll
    for (int i = 0; i < 4; ++i) { x0 += x1; x1 = rotl32(x1, RA[i]); x1 ^= x0; }
    x0 += k0;  x1 += k1 + 3u;
#pragma unroll
    for (int i = 0; i < 4; ++i) { x0 += x1; x1 = rotl32(x1, RB[i]); x1 ^= x0; }
    x0 += k1;  x1 += ks2 + 4u;
#pragma unroll
    for (int i = 0; i < 4; ++i) { x0 += x1; x1 = rotl32(x1, RA[i]); x1 ^= x0; }
    x0 += ks2; x1 += k0 + 5u;
    o0 = x0; o1 = x1;
}

__device__ __forceinline__ float erfinv_f(float u) {
    float w = -__logf((1.0f - u) * (1.0f + u));
    float p;
    if (w < 5.0f) {
        w -= 2.5f;
        p = 2.81022636e-08f;
        p = fmaf(p, w, 3.43273939e-07f);
        p = fmaf(p, w, -3.5233877e-06f);
        p = fmaf(p, w, -4.39150654e-06f);
        p = fmaf(p, w, 0.00021858087f);
        p = fmaf(p, w, -0.00125372503f);
        p = fmaf(p, w, -0.00417768164f);
        p = fmaf(p, w, 0.246640727f);
        p = fmaf(p, w, 1.50140941f);
    } else {
        w = __fsqrt_rn(w) - 3.0f;
        p = -0.000200214257f;
        p = fmaf(p, w, 0.000100950558f);
        p = fmaf(p, w, 0.00134934322f);
        p = fmaf(p, w, -0.00367342844f);
        p = fmaf(p, w, 0.00573950773f);
        p = fmaf(p, w, -0.0076224613f);
        p = fmaf(p, w, 0.00943887047f);
        p = fmaf(p, w, 1.00167406f);
        p = fmaf(p, w, 2.83297682f);
    }
    return p * u;
}

__device__ __forceinline__ float tf_normal(unsigned k) {
    unsigned r0, r1;
    threefry2x32(k, 0u, r0, r1);
    float u01 = __uint_as_float((r0 >> 9) | 0x3f800000u) - 1.0f;  // [0,1)
    float u = 2.0f * u01 - 1.0f;                                  // [-1,1)
    u = fmaxf(fminf(u, 0.9999990f), -0.9999990f);
    return 1.41421356237f * erfinv_f(u);
}

// Destination-side swizzle for A-fragments of W^T (16x32 f16 tiles).
// d = ((mb*2 + fk)*32 + lane)*16 + i  ->  source element W[kk][outc] of (64,64) W.
__device__ __forceinline__ int wt_frag_src(int d) {
    int i    = d & 15;
    int ld   = (d >> 4) & 31;
    int fk   = (d >> 9) & 1;
    int mb   = d >> 10;
    int outc = mb * 16 + (ld & 15);
    int hid  = ld >> 4;
    int g = i >> 1, j = i & 1;
    int kk = 16 * hid + i + 32 * fk;
    return kk * 64 + outc;
}

// ---------------- kernels -------------------------------------------------------

__global__ __launch_bounds__(256) void ge_zero_kernel(unsigned* __restrict__ p, int n) {
    int i = blockIdx.x * 256 + threadIdx.x;
    if (i < n) p[i] = 0u;
}

__global__ __launch_bounds__(256) void ge_hist_kernel(const int* __restrict__ gl,
                                                      int* __restrict__ counts) {
    int i = blockIdx.x * 256 + threadIdx.x;
    if (i < NB) atomicAdd(&counts[gl[i]], 1);
}

// Fused 2-layer MLP (transposed WMMA: y^T = W1^T x^T, z^T = W2^T y^T) + scatter.
// Block = 256 threads = 8 waves; each wave owns a 16-row tile (128 rows/block).
__global__ __launch_bounds__(256) void ge_mlp_segsum_kernel(
    const float* __restrict__ x, const int* __restrict__ gl,
    const float* __restrict__ W1, const float* __restrict__ b1,
    const float* __restrict__ W2, const float* __restrict__ b2,
    float* __restrict__ gsum)
{
    __shared__ _Float16 sW1s[4096];      // W1^T A-fragments, [mb][fk][lane][16]
    __shared__ _Float16 sW2s[4096];      // W2^T A-fragments
    __shared__ float    sB1[64], sB2[64];
    __shared__ _Float16 sY[8][16 * 72];  // per-wave y tile, row-major, stride 72

    const int tid = threadIdx.x;
    for (int d = tid; d < 4096; d += 256) {
        int s = wt_frag_src(d);
        sW1s[d] = (_Float16)W1[s];
        sW2s[d] = (_Float16)W2[s];
    }
    if (tid < 64) { sB1[tid] = b1[tid]; sB2[tid] = b2[tid]; }
    __syncthreads();

    const int lane = tid & 31;
    const int wave = tid >> 5;
    const int hi   = lane >> 4;   // K/column half selector
    const int nn   = lane & 15;   // reflection row within tile
    const int rowBase = blockIdx.x * 128 + wave * 16;

    int row = rowBase + nn;
    const bool vld = (row < NB);
    if (!vld) row = NB - 1;
    const int grp = gl[row];

    // B fragments of x^T straight from global: lane holds 16 contiguous f32 -> f16.
    v16h bx0, bx1;
    {
        const float* px = x + row * 64 + hi * 16;
#pragma unroll
        for (int q = 0; q < 4; ++q) {
            float4 v0 = *(const float4*)(px + 4 * q);        // cols hi*16 + 4q ..
            float4 v1 = *(const float4*)(px + 32 + 4 * q);   // cols 32 + hi*16 + 4q ..
            bx0[4 * q + 0] = (_Float16)v0.x; bx0[4 * q + 1] = (_Float16)v0.y;
            bx0[4 * q + 2] = (_Float16)v0.z; bx0[4 * q + 3] = (_Float16)v0.w;
            bx1[4 * q + 0] = (_Float16)v1.x; bx1[4 * q + 1] = (_Float16)v1.y;
            bx1[4 * q + 2] = (_Float16)v1.z; bx1[4 * q + 3] = (_Float16)v1.w;
        }
    }

    _Float16* sy = sY[wave];

    // ---- layer 1: y^T = silu(W1^T x^T + b1), packed stores into LDS ---------
#pragma unroll
    for (int mb = 0; mb < 4; ++mb) {
        v8f c;
        {
            float4 bb0 = *(const float4*)&sB1[mb * 16 + 8 * hi];
            float4 bb1 = *(const float4*)&sB1[mb * 16 + 8 * hi + 4];
            c[0] = bb0.x; c[1] = bb0.y; c[2] = bb0.z; c[3] = bb0.w;
            c[4] = bb1.x; c[5] = bb1.y; c[6] = bb1.z; c[7] = bb1.w;
        }
        v16h aw0 = *(const v16h*)&sW1s[(mb * 2 + 0) * 512 + lane * 16];
        v16h aw1 = *(const v16h*)&sW1s[(mb * 2 + 1) * 512 + lane * 16];
        c = __builtin_amdgcn_wmma_f32_16x16x32_f16(false, aw0, false, bx0, (short)0, c, false, false);
        c = __builtin_amdgcn_wmma_f32_16x16x32_f16(false, aw1, false, bx1, (short)0, c, false, false);
        v8h hh;
#pragma unroll
        for (int r = 0; r < 8; ++r) hh[r] = (_Float16)silu_f(c[r]);
        *(v8h*)&sy[nn * 72 + mb * 16 + 8 * hi] = hh;   // 8 consecutive y columns
    }
    __syncthreads();

    // B fragments of y^T: contiguous v8h pairs from this wave's LDS tile.
    v16h by0, by1;
    {
        const _Float16* py = sy + nn * 72 + hi * 16;
        v8h l0 = *(const v8h*)(py);
        v8h l1 = *(const v8h*)(py + 8);
        v8h l2 = *(const v8h*)(py + 32);
        v8h l3 = *(const v8h*)(py + 40);
        by0 = __builtin_shufflevector(l0, l1, 0, 1, 2, 3, 4, 5, 6, 7, 8, 9, 10, 11, 12, 13, 14, 15);
        by1 = __builtin_shufflevector(l2, l3, 0, 1, 2, 3, 4, 5, 6, 7, 8, 9, 10, 11, 12, 13, 14, 15);
    }

    // ---- layer 2: z^T = silu(W2^T y^T + b2), scatter-add into group sums ----
#pragma unroll
    for (int mb = 0; mb < 4; ++mb) {
        v8f c;
        {
            float4 bb0 = *(const float4*)&sB2[mb * 16 + 8 * hi];
            float4 bb1 = *(const float4*)&sB2[mb * 16 + 8 * hi + 4];
            c[0] = bb0.x; c[1] = bb0.y; c[2] = bb0.z; c[3] = bb0.w;
            c[4] = bb1.x; c[5] = bb1.y; c[6] = bb1.z; c[7] = bb1.w;
        }
        v16h aw0 = *(const v16h*)&sW2s[(mb * 2 + 0) * 512 + lane * 16];
        v16h aw1 = *(const v16h*)&sW2s[(mb * 2 + 1) * 512 + lane * 16];
        c = __builtin_amdgcn_wmma_f32_16x16x32_f16(false, aw0, false, by0, (short)0, c, false, false);
        c = __builtin_amdgcn_wmma_f32_16x16x32_f16(false, aw1, false, by1, (short)0, c, false, false);
        if (vld) {
            float* gp = &gsum[grp * 64 + mb * 16 + 8 * hi];
#pragma unroll
            for (int r = 0; r < 8; ++r) atomicAdd(gp + r, silu_f(c[r]));
        }
    }
}

// rho + heads via WMMA (h^T = W3^T gf^T) + reparameterized sample.
// Block = 256 threads = 8 waves; 16 groups per wave -> 128 groups/block.
__global__ __launch_bounds__(256) void ge_rho_kernel(
    const float* __restrict__ gsum, const int* __restrict__ counts,
    const float* __restrict__ W3, const float* __restrict__ b3,
    const float* __restrict__ w_mu, const float* __restrict__ b_mu,
    const float* __restrict__ w_lv, const float* __restrict__ b_lv,
    float* __restrict__ out_mu, float* __restrict__ out_lv,
    float* __restrict__ tau_g, float* __restrict__ ltau_g)
{
    __shared__ _Float16 sW3s[4096];
    __shared__ float sb3[64], smu[64], slv[64];

    const int tid = threadIdx.x;
    for (int d = tid; d < 4096; d += 256) sW3s[d] = (_Float16)W3[wt_frag_src(d)];
    if (tid < 64) { sb3[tid] = b3[tid]; smu[tid] = w_mu[tid]; slv[tid] = w_lv[tid]; }
    __syncthreads();

    const int lane = tid & 31;
    const int wave = tid >> 5;
    const int hi   = lane >> 4;
    const int nn   = lane & 15;
    const int kBase = blockIdx.x * 128 + wave * 16;
    const int grow  = kBase + nn;

    int cnt = counts[grow];
    const float inv = 1.0f / (float)(cnt > 0 ? cnt : 1);

    // B fragments of gf^T: 16 contiguous f32 of the group's mean row -> f16.
    v16h bg0, bg1;
    {
        const float* pg = gsum + grow * 64 + hi * 16;
#pragma unroll
        for (int q = 0; q < 4; ++q) {
            float4 v0 = *(const float4*)(pg + 4 * q);
            float4 v1 = *(const float4*)(pg + 32 + 4 * q);
            bg0[4 * q + 0] = (_Float16)(v0.x * inv); bg0[4 * q + 1] = (_Float16)(v0.y * inv);
            bg0[4 * q + 2] = (_Float16)(v0.z * inv); bg0[4 * q + 3] = (_Float16)(v0.w * inv);
            bg1[4 * q + 0] = (_Float16)(v1.x * inv); bg1[4 * q + 1] = (_Float16)(v1.y * inv);
            bg1[4 * q + 2] = (_Float16)(v1.z * inv); bg1[4 * q + 3] = (_Float16)(v1.w * inv);
        }
    }

    float pm = 0.0f, pl = 0.0f;
#pragma unroll
    for (int mb = 0; mb < 4; ++mb) {
        v8f c;
        {
            float4 bb0 = *(const float4*)&sb3[mb * 16 + 8 * hi];
            float4 bb1 = *(const float4*)&sb3[mb * 16 + 8 * hi + 4];
            c[0] = bb0.x; c[1] = bb0.y; c[2] = bb0.z; c[3] = bb0.w;
            c[4] = bb1.x; c[5] = bb1.y; c[6] = bb1.z; c[7] = bb1.w;
        }
        v16h a0 = *(const v16h*)&sW3s[(mb * 2 + 0) * 512 + lane * 16];
        v16h a1 = *(const v16h*)&sW3s[(mb * 2 + 1) * 512 + lane * 16];
        c = __builtin_amdgcn_wmma_f32_16x16x32_f16(false, a0, false, bg0, (short)0, c, false, false);
        c = __builtin_amdgcn_wmma_f32_16x16x32_f16(false, a1, false, bg1, (short)0, c, false, false);

        float4 m0 = *(const float4*)&smu[mb * 16 + 8 * hi];
        float4 m1 = *(const float4*)&smu[mb * 16 + 8 * hi + 4];
        float4 l0 = *(const float4*)&slv[mb * 16 + 8 * hi];
        float4 l1 = *(const float4*)&slv[mb * 16 + 8 * hi + 4];
        float mw[8] = {m0.x, m0.y, m0.z, m0.w, m1.x, m1.y, m1.z, m1.w};
        float lw[8] = {l0.x, l0.y, l0.z, l0.w, l1.x, l1.y, l1.z, l1.w};
#pragma unroll
        for (int r = 0; r < 8; ++r) {
            float h = silu_f(c[r]);
            pm = fmaf(h, mw[r], pm);
            pl = fmaf(h, lw[r], pl);
        }
    }
    // combine the two column halves (lanes L and L^16 share the same group).
    pm += __shfl_xor(pm, 16, 32);
    pl += __shfl_xor(pl, 16, 32);

    if (lane < 16) {
        int k = kBase + lane;
        float mu = pm + b_mu[0];
        float lv = fminf(fmaxf(pl + b_lv[0], -10.0f), 4.0f);
        float sd  = __expf(0.5f * lv);
        float eps = tf_normal((unsigned)k);
        float lt  = mu + sd * eps;
        out_mu[k] = mu;
        out_lv[k] = lv;
        tau_g[k]  = __expf(lt);
        ltau_g[k] = lt;
    }
}

__global__ __launch_bounds__(256) void ge_scatter_kernel(
    const int* __restrict__ gl, const float* __restrict__ tau_g,
    const float* __restrict__ ltau_g, float* __restrict__ out_tau,
    float* __restrict__ out_ltau)
{
    int i = blockIdx.x * 256 + threadIdx.x;
    if (i < NB) {
        int g = gl[i];
        out_tau[i]  = tau_g[g];
        out_ltau[i] = ltau_g[g];
    }
}

// ---------------- host launcher -------------------------------------------------

extern "C" void kernel_launch(void* const* d_in, const int* in_sizes, int n_in,
                              void* d_out, int out_size, void* d_ws, size_t ws_size,
                              hipStream_t stream)
{
    const float* x   = (const float*)d_in[0];
    const int*   gl  = (const int*)  d_in[1];
    const float* W1  = (const float*)d_in[2];
    const float* b1  = (const float*)d_in[3];
    const float* W2  = (const float*)d_in[4];
    const float* b2  = (const float*)d_in[5];
    const float* W3  = (const float*)d_in[6];
    const float* b3  = (const float*)d_in[7];
    const float* wmu = (const float*)d_in[8];
    const float* bmu = (const float*)d_in[9];
    const float* wlv = (const float*)d_in[10];
    const float* blv = (const float*)d_in[11];

    // workspace: gsum (NK*64 f32) | counts (NK i32) | tau_g (NK f32) | ltau_g (NK f32)
    float* gsum   = (float*)d_ws;
    int*   counts = (int*)(gsum + NK * 64);
    float* tau_g  = (float*)(counts + NK);
    float* ltau_g = tau_g + NK;

    float* out_mu   = (float*)d_out;
    float* out_lv   = out_mu + NK;
    float* out_tau  = out_lv + NK;
    float* out_ltau = out_tau + NB;

    const int nzero = NK * 64 + NK;  // gsum + counts (zero bit patterns)
    ge_zero_kernel<<<(nzero + 255) / 256, 256, 0, stream>>>((unsigned*)d_ws, nzero);
    ge_hist_kernel<<<(NB + 255) / 256, 256, 0, stream>>>(gl, counts);
    ge_mlp_segsum_kernel<<<(NB + 127) / 128, 256, 0, stream>>>(x, gl, W1, b1, W2, b2, gsum);
    ge_rho_kernel<<<NK / 128, 256, 0, stream>>>(gsum, counts, W3, b3, wmu, bmu, wlv, blv,
                                                out_mu, out_lv, tau_g, ltau_g);
    ge_scatter_kernel<<<(NB + 255) / 256, 256, 0, stream>>>(gl, tau_g, ltau_g,
                                                            out_tau, out_ltau);
}
